// RNN_4483945857342
// MI455X (gfx1250) — matrix operand order repaired and checked by reading, not compiled, not run on hardware
//
#include <hip/hip_runtime.h>
#include <hip/hip_bf16.h>
#include <cstdint>

// Problem constants
#define BB 256
#define TT 128
#define II 256
#define HH 1024
#define CC 1000

typedef __attribute__((ext_vector_type(16))) __bf16 bf16x16;
typedef __attribute__((ext_vector_type(8)))  __bf16 bf16x8;
typedef __attribute__((ext_vector_type(8)))  float  f32x8;

union ABfrag { bf16x16 v; bf16x8 h[2]; };

// ---------------------------------------------------------------------------
// CDNA5 async global->LDS copy (ASYNCcnt path). LDS operand is the low 32 bits
// of the generic pointer.
// ---------------------------------------------------------------------------
__device__ __forceinline__ void async_lds_b128(uint32_t loff, const void* gptr) {
    asm volatile("global_load_async_to_lds_b128 %0, %1, off"
                 :: "v"(loff), "v"((uint64_t)(uintptr_t)gptr) : "memory");
}

template <int N>
__device__ __forceinline__ void wait_asynccnt() {
    asm volatile("s_wait_asynccnt %0" :: "n"(N) : "memory");
}

// Raw split workgroup barrier WITHOUT the vmem-draining fence of
// __syncthreads(). Safe here: async LDS writes are drained via ASYNCcnt
// before signaling, and each wave's LDS reads are consumed (DScnt waited)
// before it signals. Lets B-fragment global loads stay in flight across it.
__device__ __forceinline__ void wg_barrier() {
    asm volatile("s_barrier_signal -1\n\ts_barrier_wait -1" ::: "memory");
}

// ---------------------------------------------------------------------------
// Pack W (N x K, row-major fp32, used as B(k,n) = W[n*K+k]) into per-fragment
// bf16 layout for v_wmma_f32_16x16x32_bf16:
//   fragment f = (k0/32)*(N/16) + (n0/16), 512 bf16 elems (1 KB)
//   elem e = lane*16 + j ; lane = hi*16 + lr ; value = B(k0 + 8*hi + (j&7) + 16*(j>>3), n0 + lr)
// ---------------------------------------------------------------------------
__global__ void pack_b_kernel(const float* __restrict__ W, __bf16* __restrict__ Bp,
                              int N, int K) {
    size_t idx = (size_t)blockIdx.x * blockDim.x + threadIdx.x;
    size_t total = (size_t)N * K;
    if (idx >= total) return;
    int    e    = (int)(idx & 511);
    size_t f    = idx >> 9;
    int lane = e >> 4, j = e & 15;
    int lr = lane & 15, hi = lane >> 4;
    int nfrag = N >> 4;
    int kb = (int)(f / nfrag), nb = (int)(f % nfrag);
    int k = kb * 32 + hi * 8 + (j & 7) + (j >> 3) * 16;
    int n = nb * 16 + lr;
    Bp[idx] = (__bf16)W[(size_t)n * K + k];
}

__global__ void bias_merge_kernel(const float* __restrict__ a,
                                  const float* __restrict__ b,
                                  float* __restrict__ o, int n) {
    int i = blockIdx.x * blockDim.x + threadIdx.x;
    if (i < n) o[i] = a[i] + b[i];
}

__global__ void f32_to_bf16_kernel(const float* __restrict__ in,
                                   __bf16* __restrict__ out, size_t n) {
    size_t i = (size_t)blockIdx.x * blockDim.x + threadIdx.x;
    if (i < n) out[i] = (__bf16)in[i];
}

__global__ void zero_bf16_kernel(__bf16* __restrict__ p, size_t n) {
    size_t i = (size_t)blockIdx.x * blockDim.x + threadIdx.x;
    if (i < n) p[i] = (__bf16)0.0f;
}

__global__ void zero_u32_kernel(unsigned* __restrict__ p, int n) {
    int i = blockIdx.x * blockDim.x + threadIdx.x;
    if (i < n) p[i] = 0u;
}

#define LDS_STRIDE 40   // bf16 elems per LDS row (80 B, keeps 16B alignment)

// ---------------------------------------------------------------------------
// Big parallel GEMM: out[M,N] = A[M,K](bf16 rm) * Bpack + bias   (fp32 out)
// Block: 128 threads = 4 waves (all in M). Tile 64(M) x 64(N); wave = 16x64.
// Triple-buffered async A staging (rotated pointers, 1 light barrier / iter)
// + register double-buffered B fragments.
// ---------------------------------------------------------------------------
__global__ __launch_bounds__(128)
void gemm_bias_kernel(const __bf16* __restrict__ A, const __bf16* __restrict__ Bp,
                      const float* __restrict__ bias, float* __restrict__ out,
                      int M, int K, int N) {
    __shared__ __bf16 lds[3][64 * LDS_STRIDE];
    const int tid  = threadIdx.x;
    const int wave = tid >> 5;
    const int lane = tid & 31;
    const int lr = lane & 15, hi = lane >> 4;
    const int bm = blockIdx.x * 64;
    const int bn = blockIdx.y * 64;
    const int nfrag = N >> 4;
    const int nkb = K >> 5;

    const int sr = tid >> 1;            // staging row 0..63
    const int sc = (tid & 1) * 16;      // staging col 0 / 16

    // Rotating triple-buffer state
    uint32_t w0 = (uint32_t)(uintptr_t)&lds[0][sr * LDS_STRIDE + sc];
    uint32_t w1 = (uint32_t)(uintptr_t)&lds[1][sr * LDS_STRIDE + sc];
    uint32_t w2 = (uint32_t)(uintptr_t)&lds[2][sr * LDS_STRIDE + sc];
    const int rb = (wave * 16 + lr) * LDS_STRIDE + 8 * hi;
    const __bf16* r0 = &lds[0][rb];
    const __bf16* r1 = &lds[1][rb];
    const __bf16* r2 = &lds[2][rb];

    const __bf16* asrc = A + (size_t)(bm + sr) * K + sc;
    const __bf16* bsrc = Bp + (((size_t)(bn >> 4)) << 9) + lane * 16;
    const size_t  bstep = (size_t)nfrag << 9;

    f32x8 acc[4] = {};
    ABfrag bc[4], bnx[4];

    // Prologue: tile 0 in flight, B frags 0 in registers
    async_lds_b128(w0, asrc);
    async_lds_b128(w0 + 16, asrc + 8);
    asrc += 32;
#pragma unroll
    for (int j = 0; j < 4; ++j) bc[j].v = *(const bf16x16*)(bsrc + (size_t)j * 512);
    bsrc += bstep;

    for (int kb = 0; kb < nkb - 1; ++kb) {
        async_lds_b128(w1, asrc);               // next A tile
        async_lds_b128(w1 + 16, asrc + 8);
        asrc += 32;
#pragma unroll
        for (int j = 0; j < 4; ++j)             // next B frags
            bnx[j].v = *(const bf16x16*)(bsrc + (size_t)j * 512);
        bsrc += bstep;

        wait_asynccnt<2>();                     // tile kb resident
        wg_barrier();                           // light barrier (no vmem drain)

        ABfrag a;
        a.h[0] = *(const bf16x8*)(r0);
        a.h[1] = *(const bf16x8*)(r0 + 16);
#pragma unroll
        for (int j = 0; j < 4; ++j)
            acc[j] = __builtin_amdgcn_wmma_f32_16x16x32_bf16(
                false, a.v, false, bc[j].v, (short)0, acc[j], false, false);
#pragma unroll
        for (int j = 0; j < 4; ++j) bc[j] = bnx[j];

        // rotate buffers: (w0,w1,w2) <- (w1,w2,w0), same for read ptrs
        uint32_t tw = w0; w0 = w1; w1 = w2; w2 = tw;
        const __bf16* tr = r0; r0 = r1; r1 = r2; r2 = tr;
    }
    wait_asynccnt<0>();
    wg_barrier();
    {
        ABfrag a;
        a.h[0] = *(const bf16x8*)(r0);
        a.h[1] = *(const bf16x8*)(r0 + 16);
#pragma unroll
        for (int j = 0; j < 4; ++j)
            acc[j] = __builtin_amdgcn_wmma_f32_16x16x32_bf16(
                false, a.v, false, bc[j].v, (short)0, acc[j], false, false);
    }

    const int m0 = bm + wave * 16 + 8 * hi;
#pragma unroll
    for (int j = 0; j < 4; ++j) {
        int n = bn + j * 16 + lr;
        float bv = bias[n];
#pragma unroll
        for (int v = 0; v < 8; ++v)
            out[(size_t)(m0 + v) * N + n] = acc[j][v] + bv;
    }
}

// ---------------------------------------------------------------------------
// Persistent fused scan: one launch runs all TT steps of one RNN layer.
//   h_{t+1} = tanh(xp[:,t,:] + h_t @ WhhT)
// Grid: dim3(BB/32, HH/64) = 8 x 16 = 128 blocks, 128 threads = 4 waves
// (2 M-waves x 2 N-waves, each wave 16x32 via 2 accumulators).
// Async triple-buffered A staging; software grid barrier between steps.
// ---------------------------------------------------------------------------
__global__ __launch_bounds__(128)
void rnn_scan_kernel(const float* __restrict__ xp, const __bf16* __restrict__ Bp,
                     __bf16* __restrict__ hA, __bf16* __restrict__ hB,
                     __bf16* __restrict__ hist, unsigned* __restrict__ bar) {
    __shared__ __bf16 lds[3][32 * LDS_STRIDE];
    const int tid   = threadIdx.x;
    const int wave  = tid >> 5;
    const int waveM = wave >> 1;          // 0..1
    const int waveN = wave & 1;           // 0..1
    const int lane  = tid & 31;
    const int lr = lane & 15, hi = lane >> 4;
    const int bm = blockIdx.x * 32;
    const int bn = blockIdx.y * 64;
    const unsigned nblk = gridDim.x * gridDim.y;
    const int nfrag = HH >> 4;
    const int nkb = HH >> 5;              // 32 K-steps

    const int sr = tid >> 2;              // staging row 0..31
    const int sc = (tid & 3) * 8;         // staging col {0,8,16,24}
    const int nb0 = (bn >> 4) + waveN * 2;

    const uint32_t lw0 = (uint32_t)(uintptr_t)&lds[0][sr * LDS_STRIDE + sc];
    const uint32_t lw1 = (uint32_t)(uintptr_t)&lds[1][sr * LDS_STRIDE + sc];
    const uint32_t lw2 = (uint32_t)(uintptr_t)&lds[2][sr * LDS_STRIDE + sc];
    const int rb = (waveM * 16 + lr) * LDS_STRIDE + 8 * hi;
    const __bf16* lr0 = &lds[0][rb];
    const __bf16* lr1 = &lds[1][rb];
    const __bf16* lr2 = &lds[2][rb];

    const __bf16* bbase0 = Bp + (((size_t)nb0) << 9) + lane * 16;
    const size_t  bstep  = (size_t)nfrag << 9;
    const size_t  arow   = (size_t)(bm + sr) * HH + sc;

    for (int t = 0; t < TT; ++t) {
        const __bf16* cur = (t & 1) ? hB : hA;
        __bf16*       nxt = (t & 1) ? hA : hB;

        uint32_t w0 = lw0, w1 = lw1, w2 = lw2;
        const __bf16 *r0 = lr0, *r1 = lr1, *r2 = lr2;
        const __bf16* asrc = cur + arow;
        const __bf16* bsrc = bbase0;

        f32x8 acc[2] = {};
        ABfrag bc0, bc1, bn0, bn1;
        async_lds_b128(w0, asrc);
        asrc += 32;
        bc0.v = *(const bf16x16*)(bsrc);
        bc1.v = *(const bf16x16*)(bsrc + 512);
        bsrc += bstep;

        for (int kb = 0; kb < nkb - 1; ++kb) {
            async_lds_b128(w1, asrc);
            asrc += 32;
            bn0.v = *(const bf16x16*)(bsrc);
            bn1.v = *(const bf16x16*)(bsrc + 512);
            bsrc += bstep;

            wait_asynccnt<1>();
            wg_barrier();

            ABfrag a;
            a.h[0] = *(const bf16x8*)(r0);
            a.h[1] = *(const bf16x8*)(r0 + 16);
            acc[0] = __builtin_amdgcn_wmma_f32_16x16x32_bf16(
                false, a.v, false, bc0.v, (short)0, acc[0], false, false);
            acc[1] = __builtin_amdgcn_wmma_f32_16x16x32_bf16(
                false, a.v, false, bc1.v, (short)0, acc[1], false, false);
            bc0 = bn0; bc1 = bn1;

            uint32_t tw = w0; w0 = w1; w1 = w2; w2 = tw;
            const __bf16* tr = r0; r0 = r1; r1 = r2; r2 = tr;
        }
        wait_asynccnt<0>();
        wg_barrier();
        {
            ABfrag a;
            a.h[0] = *(const bf16x8*)(r0);
            a.h[1] = *(const bf16x8*)(r0 + 16);
            acc[0] = __builtin_amdgcn_wmma_f32_16x16x32_bf16(
                false, a.v, false, bc0.v, (short)0, acc[0], false, false);
            acc[1] = __builtin_amdgcn_wmma_f32_16x16x32_bf16(
                false, a.v, false, bc1.v, (short)0, acc[1], false, false);
        }

        // Epilogue: h = tanh(acc + xp[b, t, :])
        const int m0 = bm + waveM * 16 + 8 * hi;
#pragma unroll
        for (int j = 0; j < 2; ++j) {
            int n = bn + waveN * 32 + j * 16 + lr;
#pragma unroll
            for (int v = 0; v < 8; ++v) {
                int m = m0 + v;                                   // batch index
                size_t xi = ((size_t)m * TT + t) * (size_t)HH + n;
                float r = tanhf(acc[j][v] + xp[xi]);
                __bf16 rb16 = (__bf16)r;
                nxt[(size_t)m * HH + n] = rb16;
                if (hist) hist[xi] = rb16;
            }
        }

        // -------- grid-wide barrier (full fence semantics required) --------
        __threadfence();                 // flush this thread's stores (agent)
        __syncthreads();                 // whole block done + fenced
        if (tid == 0) {
            __hip_atomic_fetch_add(&bar[t], 1u, __ATOMIC_RELEASE,
                                   __HIP_MEMORY_SCOPE_AGENT);
            while (__hip_atomic_load(&bar[t], __ATOMIC_ACQUIRE,
                                     __HIP_MEMORY_SCOPE_AGENT) < nblk)
                __builtin_amdgcn_s_sleep(1);
        }
        __syncthreads();                 // release all waves into step t+1
    }
}

// ---------------------------------------------------------------------------
// FC: out[b][c] = sum_h h2[b][h] * fcW[c][h] + fcb[c]   (negligible cost)
// ---------------------------------------------------------------------------
__global__ void fc_kernel(const __bf16* __restrict__ h, const float* __restrict__ Wf,
                          const float* __restrict__ bf, float* __restrict__ out) {
    int idx = blockIdx.x * blockDim.x + threadIdx.x;
    if (idx >= BB * CC) return;
    int b = idx / CC, c = idx % CC;
    const __bf16* hv = h + (size_t)b * HH;
    const float*  w  = Wf + (size_t)c * HH;
    float s = bf[c];
    for (int k = 0; k < HH; k += 8) {
        bf16x8 h8 = *(const bf16x8*)(hv + k);
        float4 w0 = *(const float4*)(w + k);
        float4 w1 = *(const float4*)(w + k + 4);
        s += (float)h8[0] * w0.x + (float)h8[1] * w0.y +
             (float)h8[2] * w0.z + (float)h8[3] * w0.w +
             (float)h8[4] * w1.x + (float)h8[5] * w1.y +
             (float)h8[6] * w1.z + (float)h8[7] * w1.w;
    }
    out[idx] = s;
}

// ---------------------------------------------------------------------------
extern "C" void kernel_launch(void* const* d_in, const int* in_sizes, int n_in,
                              void* d_out, int out_size, void* d_ws, size_t ws_size,
                              hipStream_t stream) {
    const float* x    = (const float*)d_in[0];
    const float* Wih0 = (const float*)d_in[1];
    const float* Whh0 = (const float*)d_in[2];
    const float* bih0 = (const float*)d_in[3];
    const float* bhh0 = (const float*)d_in[4];
    const float* Wih1 = (const float*)d_in[5];
    const float* Whh1 = (const float*)d_in[6];
    const float* bih1 = (const float*)d_in[7];
    const float* bhh1 = (const float*)d_in[8];
    const float* fcW  = (const float*)d_in[9];
    const float* fcb  = (const float*)d_in[10];
    float* out = (float*)d_out;

    uintptr_t p = (uintptr_t)d_ws;
    auto carve = [&](size_t bytes) -> void* {
        void* r = (void*)p;
        p += (bytes + 255) & ~(size_t)255;
        return r;
    };
    const size_t MT = (size_t)BB * TT;                  // 32768 rows
    float*    xp    = (float*)   carve(MT * HH * sizeof(float));   // 128 MB (reused)
    __bf16*   xbf   = (__bf16*)  carve(MT * II * sizeof(__bf16));  // 16 MB
    __bf16*   h1    = (__bf16*)  carve(MT * HH * sizeof(__bf16));  // 64 MB
    __bf16*   hA    = (__bf16*)  carve((size_t)BB * HH * sizeof(__bf16));
    __bf16*   hB    = (__bf16*)  carve((size_t)BB * HH * sizeof(__bf16));
    __bf16*   Wih0p = (__bf16*)  carve((size_t)HH * II * sizeof(__bf16));
    __bf16*   Whh0p = (__bf16*)  carve((size_t)HH * HH * sizeof(__bf16));
    __bf16*   Wih1p = (__bf16*)  carve((size_t)HH * HH * sizeof(__bf16));
    __bf16*   Whh1p = (__bf16*)  carve((size_t)HH * HH * sizeof(__bf16));
    float*    bias0 = (float*)   carve(HH * sizeof(float));
    float*    bias1 = (float*)   carve(HH * sizeof(float));
    unsigned* bar   = (unsigned*)carve(TT * sizeof(unsigned));

    // --- One-time weight packing / conversions ---
    pack_b_kernel<<<(HH * II + 255) / 256, 256, 0, stream>>>(Wih0, Wih0p, HH, II);
    pack_b_kernel<<<(HH * HH + 255) / 256, 256, 0, stream>>>(Whh0, Whh0p, HH, HH);
    pack_b_kernel<<<(HH * HH + 255) / 256, 256, 0, stream>>>(Wih1, Wih1p, HH, HH);
    pack_b_kernel<<<(HH * HH + 255) / 256, 256, 0, stream>>>(Whh1, Whh1p, HH, HH);
    bias_merge_kernel<<<(HH + 255) / 256, 256, 0, stream>>>(bih0, bhh0, bias0, HH);
    bias_merge_kernel<<<(HH + 255) / 256, 256, 0, stream>>>(bih1, bhh1, bias1, HH);
    f32_to_bf16_kernel<<<(unsigned)((MT * II + 255) / 256), 256, 0, stream>>>(x, xbf, MT * II);

    // --- Layer 0: xp = x @ Wih0^T + (bih0+bhh0) ---
    gemm_bias_kernel<<<dim3((unsigned)(MT / 64), HH / 64), 128, 0, stream>>>(
        xbf, Wih0p, bias0, xp, (int)MT, II, HH);

    const size_t hn = (size_t)BB * HH;
    zero_bf16_kernel<<<(unsigned)((hn + 255) / 256), 256, 0, stream>>>(hA, hn);
    zero_u32_kernel<<<1, TT, 0, stream>>>(bar, TT);
    rnn_scan_kernel<<<dim3(BB / 32, HH / 64), 128, 0, stream>>>(
        xp, Whh0p, hA, hB, h1, bar);

    // --- Layer 1: xp = h1 @ Wih1^T + (bih1+bhh1)  (reuse xp buffer) ---
    gemm_bias_kernel<<<dim3((unsigned)(MT / 64), HH / 64), 128, 0, stream>>>(
        h1, Wih1p, bias1, xp, (int)MT, HH, HH);

    zero_bf16_kernel<<<(unsigned)((hn + 255) / 256), 256, 0, stream>>>(hA, hn);
    zero_u32_kernel<<<1, TT, 0, stream>>>(bar, TT);
    rnn_scan_kernel<<<dim3(BB / 32, HH / 64), 128, 0, stream>>>(
        xp, Whh1p, hA, hB, nullptr, bar);

    // TT even: final hidden state lives in hA
    fc_kernel<<<(BB * CC + 255) / 256, 256, 0, stream>>>(hA, fcW, fcb, out);
}
